// MoE_2894807957948
// MI455X (gfx1250) — matrix operand order repaired and checked by reading, not compiled, not run on hardware
//
#include <hip/hip_runtime.h>

#define DEV __device__ __forceinline__

typedef __attribute__((ext_vector_type(16))) __bf16 v16bf;
typedef __attribute__((ext_vector_type(8)))  float  v8f;

#if defined(__has_builtin)
#if __has_builtin(__builtin_amdgcn_global_load_async_to_lds_b128) && \
    __has_builtin(__builtin_amdgcn_s_wait_asynccnt)
#define HAVE_ASYNC_LDS 1
#endif
#endif

#if defined(HAVE_ASYNC_LDS)
typedef __attribute__((__vector_size__(16))) int v4i;
typedef __attribute__((address_space(1))) v4i* gv4i_p;   // global  v4i*
typedef __attribute__((address_space(3))) v4i* lv4i_p;   // LDS     v4i*
#endif

union Frag16 {
    v16bf bf;
    uint4 q[2];
};

DEV unsigned short f2bf(float f) {
    unsigned int u = __float_as_uint(f);
    u += 0x7FFFu + ((u >> 16) & 1u);   // round-to-nearest-even
    return (unsigned short)(u >> 16);
}

// A fragment (16x32 bf16, MxK), row-major source with leading dim `ld`.
// Lane L: row = row_base + (L&15); covers K = k0..k0+7 and k0+16..k0+23,
// k0 = k_base + (L>=16 ? 8 : 0).  (ISA 7.12.2, 16-bit A 16x32 layout)
DEV v16bf load_a_gl(const unsigned short* __restrict__ W, int ld,
                    int row_base, int k_base, int lane) {
    const int r  = row_base + (lane & 15);
    const int k0 = k_base + ((lane & 16) ? 8 : 0);
    const unsigned short* p = W + (size_t)r * ld + k0;
    Frag16 f;
    f.q[0] = *(const uint4*)(p);
    f.q[1] = *(const uint4*)(p + 16);
    return f.bf;
}

// B fragment (32x16 bf16, KxN) from an LDS tile stored [col][k] row-major.
// Lane L: col = col_base + (L&15); K = k0..k0+7 and k0+16..k0+23,
// k0 = k_base + (L>=16 ? 8 : 0).
DEV v16bf load_b_lds(const unsigned short* X, int ld,
                     int col_base, int k_base, int lane) {
    const int c  = col_base + (lane & 15);
    const int k0 = k_base + ((lane & 16) ? 8 : 0);
    const unsigned short* p = X + c * ld + k0;
    Frag16 f;
    f.q[0] = *(const uint4*)(p);
    f.q[1] = *(const uint4*)(p + 16);
    return f.bf;
}

// ---------------------------------------------------------------- conversions
__global__ void cvt_bf16_kernel(const float* __restrict__ src,
                                unsigned short* __restrict__ dst, long n4) {
    long i      = (long)blockIdx.x * blockDim.x + threadIdx.x;
    long stride = (long)gridDim.x * blockDim.x;
    for (; i < n4; i += stride) {
        float4 v = ((const float4*)src)[i];
        unsigned int lo = (unsigned int)f2bf(v.x) | ((unsigned int)f2bf(v.y) << 16);
        unsigned int hi = (unsigned int)f2bf(v.z) | ((unsigned int)f2bf(v.w) << 16);
        ((uint2*)dst)[i] = make_uint2(lo, hi);
    }
}

// ---------------------------------------------------------------- gating
__global__ __launch_bounds__(256)
void gate_topk_kernel(const float* __restrict__ x,
                      const float* __restrict__ gate_w,
                      int*   __restrict__ counts,
                      int*   __restrict__ tlist,
                      float* __restrict__ wlist,
                      int N) {
    constexpr int D = 512, E = 8;
    __shared__ float g[E * D];
    for (int i = threadIdx.x; i < E * D; i += blockDim.x) g[i] = gate_w[i];
    __syncthreads();

    const int n = blockIdx.x * blockDim.x + threadIdx.x;
    if (n >= N) return;

    const float4* xr = (const float4*)(x + (size_t)n * D);
    float acc[E];
#pragma unroll
    for (int e = 0; e < E; ++e) acc[e] = 0.0f;
    for (int d4 = 0; d4 < D / 4; ++d4) {
        float4 xv = xr[d4];
#pragma unroll
        for (int e = 0; e < E; ++e) {
            float4 gv = *(const float4*)(g + e * D + d4 * 4);
            acc[e] += xv.x * gv.x + xv.y * gv.y + xv.z * gv.z + xv.w * gv.w;
        }
    }
    int i0 = 0; float l0 = acc[0];
#pragma unroll
    for (int e = 1; e < E; ++e) if (acc[e] > l0) { l0 = acc[e]; i0 = e; }
    int i1 = -1; float l1 = -3.4e38f;
#pragma unroll
    for (int e = 0; e < E; ++e) if (e != i0 && acc[e] > l1) { l1 = acc[e]; i1 = e; }

    float e1 = __expf(l1 - l0);          // <= 1
    float s  = 1.0f / (1.0f + e1);
    float w0 = s, w1v = e1 * s;

    int s0 = atomicAdd(&counts[i0], 1);
    tlist[(size_t)i0 * N + s0] = n;  wlist[(size_t)i0 * N + s0] = w0;
    int s1 = atomicAdd(&counts[i1], 1);
    tlist[(size_t)i1 * N + s1] = n;  wlist[(size_t)i1 * N + s1] = w1v;
}

// ---------------------------------------------------------------- fused FFN
// grid = (E, ceil(N/64)); block = 256 threads = 8 wave32s.
// Stage 1: wave w owns j-slice [w*192, w*192+192); sweeps all 4 token groups,
//          reusing each A (w1/w3) fragment 4x. Weights read once per block.
// Stage 2: wave w owns d-slice [w*64, w*64+64); 16 persistent accumulators
//          (4 d-tiles x 4 token groups) reuse both A and B fragments 4x.
__global__ __launch_bounds__(256, 1)
void moe_ffn_kernel(const unsigned short* __restrict__ xb,
                    const unsigned short* __restrict__ w1b,
                    const unsigned short* __restrict__ w3b,
                    const unsigned short* __restrict__ w2b,
                    const int*   __restrict__ counts,
                    const int*   __restrict__ tlist,
                    const float* __restrict__ wlist,
                    float* __restrict__ out,
                    int N) {
    constexpr int D = 512, H = 1536;
    constexpr int LDX = D + 8;    // pad: break LDS bank conflicts
    constexpr int LDH = H + 8;
    __shared__ unsigned short x_lds[64 * LDX];   //  66,560 B
    __shared__ unsigned short h_lds[64 * LDH];   // 197,632 B
    __shared__ int   toks[64];
    __shared__ float gws[64];

    const int e    = blockIdx.x;
    const int tile = blockIdx.y;
    const int cnt  = counts[e];
    if (tile * 64 >= cnt) return;    // uniform exit, before any barrier

    const int tid  = threadIdx.x;
    const int lane = tid & 31;
    const int wave = tid >> 5;

    if (tid < 64) {
        int slot = tile * 64 + tid;
        if (slot < cnt) {
            toks[tid] = tlist[(size_t)e * N + slot];
            gws[tid]  = wlist[(size_t)e * N + slot];
        } else {
            toks[tid] = 0;       // padded lane: weight 0 -> contributes nothing
            gws[tid]  = 0.0f;
        }
    }
    __syncthreads();

    // Gather 64-token x tile (bf16) into LDS: thread -> (row, quarter-row).
    {
        const int r = tid >> 2;
        const int q = tid & 3;
        const unsigned short* src = xb + (size_t)toks[r] * D + q * 128;
        unsigned short* dst = x_lds + (size_t)r * LDX + q * 128;
#if defined(HAVE_ASYNC_LDS)
#pragma unroll
        for (int i = 0; i < 16; ++i)
            __builtin_amdgcn_global_load_async_to_lds_b128(
                (gv4i_p)(src + i * 8), (lv4i_p)(dst + i * 8), 0, 0);
        __builtin_amdgcn_s_wait_asynccnt(0);
#else
#pragma unroll
        for (int i = 0; i < 16; ++i)
            *(uint4*)(dst + i * 8) = ((const uint4*)src)[i];
#endif
    }
    __syncthreads();

    // ---- Stage 1: h = silu(x @ w1^T) * (x @ w3^T)  -> LDS (bf16)
    const unsigned short* W1 = w1b + (size_t)e * H * D;
    const unsigned short* W3 = w3b + (size_t)e * H * D;
    const v8f vzero = {};
    for (int jt = 0; jt < 12; ++jt) {
        const int jbase = wave * (H / 8) + jt * 16;
        if (jt + 1 < 12) {
            __builtin_prefetch(W1 + (size_t)(jbase + 16) * D, 0, 0);
            __builtin_prefetch(W3 + (size_t)(jbase + 16) * D, 0, 0);
        }
        v8f acc1[4], acc3[4];
#pragma unroll
        for (int tg = 0; tg < 4; ++tg) { acc1[tg] = vzero; acc3[tg] = vzero; }

#pragma unroll 2
        for (int kt = 0; kt < D / 32; ++kt) {
            const int kb = kt * 32;
            v16bf a1 = load_a_gl(W1, D, jbase, kb, lane);
            v16bf a3 = load_a_gl(W3, D, jbase, kb, lane);
#pragma unroll
            for (int tg = 0; tg < 4; ++tg) {
                v16bf b = load_b_lds(x_lds, LDX, tg * 16, kb, lane);
                acc1[tg] = __builtin_amdgcn_wmma_f32_16x16x32_bf16(
                    false, a1, false, b, (short)0, acc1[tg], false, false);
                acc3[tg] = __builtin_amdgcn_wmma_f32_16x16x32_bf16(
                    false, a3, false, b, (short)0, acc3[tg], false, false);
            }
        }
        // D-frag: lane holds rows j = jbase + (L>=16?8:0) + r, col = token.
        const int joff = jbase + ((lane & 16) ? 8 : 0);
#pragma unroll
        for (int tg = 0; tg < 4; ++tg) {
            const int tl = tg * 16 + (lane & 15);
            uint4 pk;
            unsigned int wds[4];
#pragma unroll
            for (int r = 0; r < 4; ++r) {
                float g0 = acc1[tg][2 * r], g1 = acc1[tg][2 * r + 1];
                float h0 = (g0 / (1.0f + __expf(-g0))) * acc3[tg][2 * r];
                float h1 = (g1 / (1.0f + __expf(-g1))) * acc3[tg][2 * r + 1];
                wds[r] = (unsigned int)f2bf(h0) | ((unsigned int)f2bf(h1) << 16);
            }
            pk.x = wds[0]; pk.y = wds[1]; pk.z = wds[2]; pk.w = wds[3];
            *(uint4*)(h_lds + tl * LDH + joff) = pk;
        }
    }
    __syncthreads();

    // ---- Stage 2: y = h @ w2^T, scatter-add scaled by gate weight
    const unsigned short* W2 = w2b + (size_t)e * D * H;
    const int dslice = wave * (D / 8);           // 64-wide d' slice per wave
    v8f acc[4][4];                               // [d-tile][token-group]
#pragma unroll
    for (int dt = 0; dt < 4; ++dt)
#pragma unroll
        for (int tg = 0; tg < 4; ++tg) acc[dt][tg] = vzero;

    for (int jt = 0; jt < H / 32; ++jt) {
        const int kb = jt * 32;
        v16bf b[4];
#pragma unroll
        for (int tg = 0; tg < 4; ++tg)
            b[tg] = load_b_lds(h_lds, LDH, tg * 16, kb, lane);
#pragma unroll
        for (int dt = 0; dt < 4; ++dt) {
            v16bf a = load_a_gl(W2, H, dslice + dt * 16, kb, lane);
#pragma unroll
            for (int tg = 0; tg < 4; ++tg)
                acc[dt][tg] = __builtin_amdgcn_wmma_f32_16x16x32_bf16(
                    false, a, false, b[tg], (short)0, acc[dt][tg], false, false);
        }
    }

    int   tokr[4];
    float gwr[4];
#pragma unroll
    for (int tg = 0; tg < 4; ++tg) {
        tokr[tg] = toks[tg * 16 + (lane & 15)];
        gwr[tg]  = gws[tg * 16 + (lane & 15)];
    }
#pragma unroll
    for (int dt = 0; dt < 4; ++dt) {
        const int doff = dslice + dt * 16 + ((lane & 16) ? 8 : 0);
#pragma unroll
        for (int tg = 0; tg < 4; ++tg) {
            float* orow = out + (size_t)tokr[tg] * D + doff;
#pragma unroll
            for (int r = 0; r < 8; ++r)
                atomicAdd(orow + r, acc[dt][tg][r] * gwr[tg]);
        }
    }
}

// ---------------------------------------------------------------- launch
extern "C" void kernel_launch(void* const* d_in, const int* in_sizes, int n_in,
                              void* d_out, int out_size, void* d_ws, size_t ws_size,
                              hipStream_t stream) {
    (void)n_in; (void)ws_size;
    const float* x      = (const float*)d_in[0];
    const float* gate_w = (const float*)d_in[1];
    const float* w1     = (const float*)d_in[2];
    const float* w3     = (const float*)d_in[3];
    const float* w2     = (const float*)d_in[4];
    float* out = (float*)d_out;

    const int  E = 8;
    const int  D = in_sizes[1] / E;            // 512
    const long N = (long)in_sizes[0] / D;      // 32768
    const int  H = (int)((long)in_sizes[2] / ((long)E * D));  // 1536

    // workspace layout (all 256B-aligned by construction)
    char* p = (char*)d_ws;
    unsigned short* xb  = (unsigned short*)p; p += (size_t)N * D * 2;
    unsigned short* w1b = (unsigned short*)p; p += (size_t)E * H * D * 2;
    unsigned short* w3b = (unsigned short*)p; p += (size_t)E * H * D * 2;
    unsigned short* w2b = (unsigned short*)p; p += (size_t)E * H * D * 2;
    int*   counts = (int*)p;   p += 256;
    int*   tlist  = (int*)p;   p += (size_t)E * N * 4;
    float* wlist  = (float*)p; p += (size_t)E * N * 4;

    (void)hipMemsetAsync(counts, 0, 256, stream);
    (void)hipMemsetAsync(out, 0, (size_t)out_size * sizeof(float), stream);

    // fp32 -> bf16 conversions
    {
        long n4;
        n4 = (long)N * D / 4;
        cvt_bf16_kernel<<<2048, 256, 0, stream>>>(x,  xb,  n4);
        n4 = (long)E * H * D / 4;
        cvt_bf16_kernel<<<2048, 256, 0, stream>>>(w1, w1b, n4);
        cvt_bf16_kernel<<<2048, 256, 0, stream>>>(w3, w3b, n4);
        cvt_bf16_kernel<<<2048, 256, 0, stream>>>(w2, w2b, n4);
    }

    // gating + routing
    gate_topk_kernel<<<dim3((unsigned)((N + 255) / 256)), 256, 0, stream>>>(
        x, gate_w, counts, tlist, wlist, (int)N);

    // fused expert FFN (grid covers worst-case routing)
    dim3 grid((unsigned)E, (unsigned)((N + 63) / 64));
    moe_ffn_kernel<<<grid, 256, 0, stream>>>(xb, w1b, w3b, w2b,
                                             counts, tlist, wlist, out, (int)N);
}
